// TaggingProjector_30605936951493
// MI455X (gfx1250) — matrix operand order repaired and checked by reading, not compiled, not run on hardware
//
#include <hip/hip_runtime.h>

// ---------------------------------------------------------------------------
// TaggingProjector for MI455X (gfx1250, wave32, WMMA)
//
// logits = hidden(66048x768) @ W^T(768x72) + b ; softmax over groups of 3.
// Memory-bound problem (203MB hidden stream, 23.3 TB/s -> ~9.5us floor).
// fp32 multiplies would be compute-bound, so we emulate fp32 with a
// bf16 hi/lo split and 3x v_wmma_f32_16x16x32_bf16 products per tile,
// accumulating in fp32 -- ~24 GFLOP bf16, well under the HBM roof.
// ---------------------------------------------------------------------------

typedef __attribute__((ext_vector_type(16))) __bf16 v16bf;
typedef __attribute__((ext_vector_type(8)))  float  v8f;
typedef __attribute__((ext_vector_type(4)))  float  v4f;

namespace {
constexpr int kH       = 768;       // hidden size (K)
constexpr int kB       = 8;
constexpr int kL       = 8256;
constexpr int kR       = 24;
constexpr int kNCols   = 72;        // R*3
constexpr int kNT      = 5;         // column tiles of 16 (72 -> 80, padded)
constexpr int kKSteps  = kH / 32;   // 24 K-steps of 32
constexpr int kLPad    = 84;        // LDS row stride (pad 80 -> 84, bank-conflict dodge)
constexpr int kWaves   = 4;         // waves per block
constexpr int kRowsPW  = 32;        // rows per wave = two 16-row M tiles
}

// ---------------------------------------------------------------------------
// Prep: split W (72x768 fp32, row-major [n][h]) into bf16 hi/lo planes laid
// out in exact WMMA B-fragment order:
//   frag(plane, kk, nt) = 32 lanes x 8 dwords (16 bf16 / lane)
//   lane l -> column n = nt*16 + (l&15), k = kk*32 + (l>>4)*8 + e + (e>=8 ? 8 : 0)  (e=0..15)
// One thread produces one packed dword (2 bf16). Total 61440 threads.
// ---------------------------------------------------------------------------
__global__ void tp_prep_w(const float* __restrict__ W, __bf16* __restrict__ wsb) {
  int idx = blockIdx.x * 256 + threadIdx.x;        // exact: 240*256 = 61440
  int j    = idx & 7;
  int t    = idx >> 3;
  int lane = t & 31;  t >>= 5;
  int nt   = t % kNT; t /= kNT;
  int kk   = t % kKSteps;
  int plane = t / kKSteps;                         // 0 = hi, 1 = lo residual
  int n  = nt * 16 + (lane & 15); n = n > (kNCols - 1) ? (kNCols - 1) : n;
  int kb = kk * 32 + (lane >> 4) * 8 + 2 * j + (j >= 4 ? 8 : 0);
  float w0 = W[n * kH + kb];
  float w1 = W[n * kH + kb + 1];
  if (plane) {
    w0 -= (float)(__bf16)w0;                       // residual after bf16 round
    w1 -= (float)(__bf16)w1;
  }
  wsb[2 * idx + 0] = (__bf16)w0;
  wsb[2 * idx + 1] = (__bf16)w1;
}

// Split 16 fp32 values (already in A-fragment element order) into bf16 hi/lo.
__device__ inline void split16(const v4f x0, const v4f x1, const v4f x2,
                               const v4f x3, v16bf& hi, v16bf& lo) {
  float x[16];
#pragma unroll
  for (int e = 0; e < 4; ++e) {
    x[e] = x0[e]; x[4 + e] = x1[e]; x[8 + e] = x2[e]; x[12 + e] = x3[e];
  }
#pragma unroll
  for (int e = 0; e < 16; ++e) {
    __bf16 h = (__bf16)x[e];
    hi[e] = h;
    lo[e] = (__bf16)(x[e] - (float)h);
  }
}

// ---------------------------------------------------------------------------
// Main: per wave, 2 M-tiles (32 rows) x 5 N-tiles; 24 K-steps of 32.
// A fragment layout (16-bit A 16x32, wave32): lane l holds row m = l&15,
// K = (l>>4)*8 + {0..7} and +16..+23  -> two pairs of contiguous float4s.
// ---------------------------------------------------------------------------
__global__ void __launch_bounds__(kWaves * 32)
tp_gemm_softmax(const float* __restrict__ X, const v16bf* __restrict__ Bf,
                const float* __restrict__ bias, float* __restrict__ out) {
  __shared__ float lds[kWaves * kRowsPW * kLPad];

  const int lane    = threadIdx.x & 31;
  const int wave    = threadIdx.x >> 5;
  const int mrow    = lane & 15;
  const int half    = lane >> 4;
  const int rowBase = (blockIdx.x * kWaves + wave) * kRowsPW;

  const float* a0 = X + (size_t)(rowBase + mrow) * kH + half * 8;
  const float* a1 = a0 + (size_t)16 * kH;

  v8f acc0[kNT], acc1[kNT];
#pragma unroll
  for (int nt = 0; nt < kNT; ++nt)
#pragma unroll
    for (int e = 0; e < 8; ++e) { acc0[nt][e] = 0.f; acc1[nt][e] = 0.f; }

  for (int kk = 0; kk < kKSteps; ++kk) {
    const float* p0 = a0 + kk * 32;
    const float* p1 = a1 + kk * 32;
    if (kk + 1 < kKSteps) {                        // global_prefetch next K-step
      __builtin_prefetch(p0 + 32, 0, 0);
      __builtin_prefetch(p1 + 32, 0, 0);
    }
    // Streaming 203MB once: nontemporal b128 loads, full cacheline use.
    v4f x00 = __builtin_nontemporal_load((const v4f*)(p0));
    v4f x01 = __builtin_nontemporal_load((const v4f*)(p0 + 4));
    v4f x02 = __builtin_nontemporal_load((const v4f*)(p0 + 16));
    v4f x03 = __builtin_nontemporal_load((const v4f*)(p0 + 20));
    v4f x10 = __builtin_nontemporal_load((const v4f*)(p1));
    v4f x11 = __builtin_nontemporal_load((const v4f*)(p1 + 4));
    v4f x12 = __builtin_nontemporal_load((const v4f*)(p1 + 16));
    v4f x13 = __builtin_nontemporal_load((const v4f*)(p1 + 20));

    v16bf A0h, A0l, A1h, A1l;
    split16(x00, x01, x02, x03, A0h, A0l);
    split16(x10, x11, x12, x13, A1h, A1l);

#pragma unroll
    for (int nt = 0; nt < kNT; ++nt) {
      v16bf Bh = Bf[(size_t)(kk * kNT + nt) * 32 + lane];
      v16bf Bl = Bf[(size_t)((kKSteps + kk) * kNT + nt) * 32 + lane];
      // fp32-emulating accumulation: AhBh + AlBh + AhBl (drop AlBl).
      acc0[nt] = __builtin_amdgcn_wmma_f32_16x16x32_bf16(false, A0h, false, Bh,
                                                         (short)0, acc0[nt], false, false);
      acc1[nt] = __builtin_amdgcn_wmma_f32_16x16x32_bf16(false, A1h, false, Bh,
                                                         (short)0, acc1[nt], false, false);
      acc0[nt] = __builtin_amdgcn_wmma_f32_16x16x32_bf16(false, A0l, false, Bh,
                                                         (short)0, acc0[nt], false, false);
      acc1[nt] = __builtin_amdgcn_wmma_f32_16x16x32_bf16(false, A1l, false, Bh,
                                                         (short)0, acc1[nt], false, false);
      acc0[nt] = __builtin_amdgcn_wmma_f32_16x16x32_bf16(false, A0h, false, Bl,
                                                         (short)0, acc0[nt], false, false);
      acc1[nt] = __builtin_amdgcn_wmma_f32_16x16x32_bf16(false, A1h, false, Bl,
                                                         (short)0, acc1[nt], false, false);
    }
  }

  // ---- epilogue: bias, stash logits (C layout: lane -> col, VGPR j -> row) --
  float* slab = lds + wave * (kRowsPW * kLPad);
#pragma unroll
  for (int nt = 0; nt < kNT; ++nt) {
    int n = nt * 16 + mrow;
    float bv = bias[n > (kNCols - 1) ? (kNCols - 1) : n];
    int col = nt * 16 + mrow;
#pragma unroll
    for (int j = 0; j < 8; ++j) {
      int r0 = half * 8 + j;
      slab[r0 * kLPad + col]        = acc0[nt][j] + bv;
      slab[(r0 + 16) * kLPad + col] = acc1[nt][j] + bv;
    }
  }
  __syncthreads();

  // ---- softmax over t (3) and scatter to (B, R, L, 3) ----------------------
#pragma unroll 1
  for (int g = lane; g < kRowsPW * kR; g += 32) {   // 768 groups / wave, exact
    int row = g / kR;
    int r   = g - row * kR;
    const float* z = slab + row * kLPad + r * 3;
    float z0 = z[0], z1 = z[1], z2 = z[2];
    float m  = fmaxf(z0, fmaxf(z1, z2));
    float e0 = __expf(z0 - m), e1 = __expf(z1 - m), e2 = __expf(z2 - m);
    float inv = 1.0f / (e0 + e1 + e2);
    int gr = rowBase + row;
    int bb = gr / kL;
    int ll = gr - bb * kL;
    size_t o = ((size_t)(bb * kR + r) * kL + ll) * 3;
    __builtin_nontemporal_store(e0 * inv, out + o + 0);
    __builtin_nontemporal_store(e1 * inv, out + o + 1);
    __builtin_nontemporal_store(e2 * inv, out + o + 2);
  }
}

extern "C" void kernel_launch(void* const* d_in, const int* in_sizes, int n_in,
                              void* d_out, int out_size, void* d_ws, size_t ws_size,
                              hipStream_t stream) {
  const float* hidden = (const float*)d_in[0];   // (8, 8256, 768) fp32
  const float* W      = (const float*)d_in[1];   // (24, 3, 768)   fp32
  const float* bias   = (const float*)d_in[2];   // (24, 3)        fp32
  float* out          = (float*)d_out;           // (8, 24, 8256, 3) fp32
  __bf16* wsb         = (__bf16*)d_ws;           // 245,760 B of fragment-packed W

  // 1) split+pack W into WMMA B-fragment order (hi/lo bf16 planes)
  tp_prep_w<<<240, 256, 0, stream>>>(W, wsb);

  // 2) fused GEMM (bf16x3 WMMA) + bias + softmax
  //    66048 rows / (4 waves * 32 rows) = 516 blocks, exact.
  tp_gemm_softmax<<<516, kWaves * 32, 0, stream>>>(
      hidden, (const v16bf*)d_ws, bias, out);
}
